// GlobalAttentionNet_85452669321992
// MI455X (gfx1250) — matrix operand order repaired and checked by reading, not compiled, not run on hardware
//
#include <hip/hip_runtime.h>
#include <hip/hip_bf16.h>
#include <math.h>

typedef __attribute__((ext_vector_type(16))) __bf16 v16bf;
typedef __attribute__((ext_vector_type(8)))  float  v8f;

#define HID 128

// ---------------------------------------------------------------- utilities
__global__ void zero_f32(float* __restrict__ p, long long n) {
    long long i = (long long)blockIdx.x * blockDim.x + threadIdx.x;
    if (i < n) p[i] = 0.0f;
}

__global__ void count_deg(const int* __restrict__ ei, int* __restrict__ cnt, int n_edges) {
    int e = blockIdx.x * blockDim.x + threadIdx.x;
    if (e < n_edges) atomicAdd(&cnt[ei[n_edges + e]], 1);
}

__global__ void inv_deg(const int* __restrict__ cnt, float* __restrict__ inv, int n) {
    int i = blockIdx.x * blockDim.x + threadIdx.x;
    if (i < n) inv[i] = 1.0f / fmaxf((float)cnt[i], 1.0f);
}

// ------------------------------------------------- weight pre-pack into WMMA
// B fragment layout used here (bf16 32x16 tile, wave32; contiguous k, same order as A):
//   lane L holds column N = nt*16 + (L&15); half h = L>>4; element e -> K = kc*32 + h*16 + e
__global__ void pack_b(const float* __restrict__ W, __bf16* __restrict__ out) {
    int tid = blockIdx.x * blockDim.x + threadIdx.x;   // 4 ktiles * 8 ntiles * 32 lanes
    if (tid >= 4 * 8 * 32) return;
    int lane = tid & 31;
    int nt   = (tid >> 5) & 7;
    int kc   = tid >> 8;
    int n    = nt * 16 + (lane & 15);
    int kb   = kc * 32 + (lane >> 4) * 16;
    __bf16* o = out + (long long)tid * 16;
    #pragma unroll
    for (int e = 0; e < 16; ++e) o[e] = (__bf16)W[(kb + e) * HID + n];
}

// ---------------------------------------------------------- edge scatter-add
// one wave per edge; lane covers 4 contiguous features (coalesced 512B read)
__global__ void scatter_edges(const float* __restrict__ h, const int* __restrict__ ei,
                              float* __restrict__ msg, int n_edges) {
    long long tid = (long long)blockIdx.x * blockDim.x + threadIdx.x;
    int e = (int)(tid >> 5);
    if (e >= n_edges) return;
    int lane = (int)(tid & 31);
    int src = ei[e];
    int dst = ei[n_edges + e];
    const float4 v = *(const float4*)(h + (long long)src * HID + lane * 4);
    float* p = msg + (long long)dst * HID + lane * 4;
    atomicAdd(p + 0, v.x);
    atomicAdd(p + 1, v.y);
    atomicAdd(p + 2, v.z);
    atomicAdd(p + 3, v.w);
}

// ---------------------------------------------------------------- WMMA GEMM
// DUAL=1: out[M x 128] = relu( (A1 * inv[row]) @ B1 + A2 @ B2 + bias )   (SAGE layer)
// DUAL=0: out[M x 128] = relu(  A1            @ B1            + bias )   (lin1)
// One wave computes a 16-row x 128-col strip. All lane divergence is wave-uniform,
// EXEC is all-1s at every WMMA.
template <bool DUAL>
__global__ void sage_gemm(const float* __restrict__ A1, const float* __restrict__ invc,
                          const float* __restrict__ A2,
                          const __bf16* __restrict__ B1p, const __bf16* __restrict__ B2p,
                          const float* __restrict__ bias, float* __restrict__ out,
                          int M, int do_relu) {
    int wave = threadIdx.x >> 5;
    int lane = threadIdx.x & 31;
    int rowBlk = blockIdx.x * 4 + wave;           // wave-uniform
    int rowBase = rowBlk * 16;
    if (rowBase >= M) return;                     // whole-wave exit, EXEC stays full

    int mrow = rowBase + (lane & 15);
    int half = lane >> 4;
    float scale = DUAL ? invc[mrow] : 1.0f;

    v8f acc[8];
    #pragma unroll
    for (int t = 0; t < 8; ++t) acc[t] = (v8f){0.f,0.f,0.f,0.f,0.f,0.f,0.f,0.f};

    #pragma unroll
    for (int kc = 0; kc < 4; ++kc) {
        // A fragment: contiguous k order, same as packed B. element e -> K = kc*32 + half*16 + e
        v16bf a1 = {}, a2 = {};
        const float* r1 = A1 + (long long)mrow * HID + kc * 32 + half * 16;
        #pragma unroll
        for (int e = 0; e < 8; ++e) a1[e]     = (__bf16)(r1[e]      * scale);
        #pragma unroll
        for (int e = 0; e < 8; ++e) a1[8 + e] = (__bf16)(r1[8 + e] * scale);
        if (DUAL) {
            const float* r2 = A2 + (long long)mrow * HID + kc * 32 + half * 16;
            #pragma unroll
            for (int e = 0; e < 8; ++e) a2[e]     = (__bf16)r2[e];
            #pragma unroll
            for (int e = 0; e < 8; ++e) a2[8 + e] = (__bf16)r2[8 + e];
        }
        #pragma unroll
        for (int nt = 0; nt < 8; ++nt) {
            v16bf b1 = *(const v16bf*)(B1p + ((long long)((kc * 8 + nt) * 32 + lane) * 16));
            acc[nt] = __builtin_amdgcn_wmma_f32_16x16x32_bf16(
                false, a1, false, b1, (short)0, acc[nt], false, false);
            if (DUAL) {
                v16bf b2 = *(const v16bf*)(B2p + ((long long)((kc * 8 + nt) * 32 + lane) * 16));
                acc[nt] = __builtin_amdgcn_wmma_f32_16x16x32_bf16(
                    false, a2, false, b2, (short)0, acc[nt], false, false);
            }
        }
    }
    // C/D layout: VGPR r -> row = rowBase + r + 8*half; col = nt*16 + (lane&15)
    #pragma unroll
    for (int nt = 0; nt < 8; ++nt) {
        int col = nt * 16 + (lane & 15);
        float bv = bias[col];
        #pragma unroll
        for (int r = 0; r < 8; ++r) {
            int row = rowBase + r + 8 * half;
            float v = acc[nt][r] + bv;
            if (do_relu) v = fmaxf(v, 0.0f);
            out[(long long)row * HID + col] = v;
        }
    }
}

// ------------------------------------------------------------- attention pool
__device__ inline unsigned f2key(float x) {
    unsigned u = __float_as_uint(x);
    return (u & 0x80000000u) ? ~u : (u | 0x80000000u);
}
__device__ inline float key2f(unsigned k) {
    unsigned u = (k & 0x80000000u) ? (k & 0x7FFFFFFFu) : ~k;
    return __uint_as_float(u);
}

__global__ void gate_kernel(const float* __restrict__ h, const float* __restrict__ gw,
                            const float* __restrict__ gb, const int* __restrict__ batch,
                            float* __restrict__ gate, unsigned* __restrict__ mkey, int n) {
    long long tid = (long long)blockIdx.x * blockDim.x + threadIdx.x;
    int node = (int)(tid >> 5);
    if (node >= n) return;
    int lane = (int)(tid & 31);
    float4 v = *(const float4*)(h + (long long)node * HID + lane * 4);
    float4 w = *(const float4*)(gw + lane * 4);
    float p = v.x * w.x + v.y * w.y + v.z * w.z + v.w * w.w;
    #pragma unroll
    for (int off = 16; off; off >>= 1) p += __shfl_xor(p, off, 32);
    if (lane == 0) {
        float g = p + gb[0];
        gate[node] = g;
        atomicMax(&mkey[batch[node]], f2key(g));
    }
}

__global__ void att_accum(const float* __restrict__ h, const float* __restrict__ gate,
                          const int* __restrict__ batch, const unsigned* __restrict__ mkey,
                          float* __restrict__ s, float* __restrict__ pooledNum, int n) {
    long long tid = (long long)blockIdx.x * blockDim.x + threadIdx.x;
    int node = (int)(tid >> 5);
    if (node >= n) return;
    int lane = (int)(tid & 31);
    int g = batch[node];
    float e = __expf(gate[node] - key2f(mkey[g]));
    if (lane == 0) atomicAdd(&s[g], e);
    float4 v = *(const float4*)(h + (long long)node * HID + lane * 4);
    float* p = pooledNum + (long long)g * HID + lane * 4;
    atomicAdd(p + 0, e * v.x);
    atomicAdd(p + 1, e * v.y);
    atomicAdd(p + 2, e * v.z);
    atomicAdd(p + 3, e * v.w);
}

__global__ void pooled_div(const float* __restrict__ num, const float* __restrict__ s,
                           float* __restrict__ pooled, int total) {
    int i = blockIdx.x * blockDim.x + threadIdx.x;
    if (i < total) pooled[i] = num[i] / s[i >> 7];
}

// ------------------------------------------------------- lin2 + log_softmax
__global__ void head_kernel(const float* __restrict__ h1, const float* __restrict__ w2,
                            const float* __restrict__ b2, float* __restrict__ out,
                            int n_classes) {
    __shared__ float sh[HID];
    __shared__ float so[16];
    int g = blockIdx.x;
    int t = threadIdx.x;
    sh[t] = h1[(long long)g * HID + t];
    __syncthreads();
    if (t < n_classes) {
        float a = b2[t];
        for (int k = 0; k < HID; ++k) a += sh[k] * w2[k * n_classes + t];
        so[t] = a;
    }
    __syncthreads();
    if (t == 0) {
        float mx = -INFINITY;
        for (int i = 0; i < n_classes; ++i) mx = fmaxf(mx, so[i]);
        float se = 0.0f;
        for (int i = 0; i < n_classes; ++i) se += __expf(so[i] - mx);
        float lse = mx + __logf(se);
        for (int i = 0; i < n_classes; ++i) out[(long long)g * n_classes + i] = so[i] - lse;
    }
}

// ---------------------------------------------------------------- launcher
extern "C" void kernel_launch(void* const* d_in, const int* in_sizes, int n_in,
                              void* d_out, int out_size, void* d_ws, size_t ws_size,
                              hipStream_t stream) {
    const float* x       = (const float*)d_in[0];
    const int*   ei      = (const int*)  d_in[1];
    const int*   batch   = (const int*)  d_in[2];
    const float* w1l     = (const float*)d_in[3];
    const float* b1      = (const float*)d_in[4];
    const float* w1r     = (const float*)d_in[5];
    const float* w2l     = (const float*)d_in[6];
    const float* b2      = (const float*)d_in[7];
    const float* w2r     = (const float*)d_in[8];
    const float* w3l     = (const float*)d_in[9];
    const float* b3      = (const float*)d_in[10];
    const float* w3r     = (const float*)d_in[11];
    const float* gate_w  = (const float*)d_in[12];
    const float* gate_b  = (const float*)d_in[13];
    const float* lin1_w  = (const float*)d_in[14];
    const float* lin1_b  = (const float*)d_in[15];
    const float* lin2_w  = (const float*)d_in[16];
    const float* lin2_b  = (const float*)d_in[17];

    const int n_nodes   = in_sizes[0] / HID;
    const int n_edges   = in_sizes[1] / 2;
    const int n_classes = in_sizes[17];          // lin2_b length
    const int n_graphs  = out_size / n_classes;

    // ---- workspace carve (256B aligned) ----
    char* ws = (char*)d_ws;
    size_t off = 0;
    auto carve = [&](size_t bytes) -> char* {
        char* p = ws + off;
        off += (bytes + 255) & ~(size_t)255;
        return p;
    };
    const size_t nodeMat = (size_t)n_nodes * HID * sizeof(float);
    float*    hA     = (float*)carve(nodeMat);
    float*    hB     = (float*)carve(nodeMat);
    float*    msg    = (float*)carve(nodeMat);
    int*      cnt    = (int*)  carve((size_t)n_nodes * sizeof(int));
    float*    inv    = (float*)carve((size_t)n_nodes * sizeof(float));
    float*    gate   = (float*)carve((size_t)n_nodes * sizeof(float));
    __bf16*   pw1l   = (__bf16*)carve((size_t)HID * HID * 2);
    __bf16*   pw1r   = (__bf16*)carve((size_t)HID * HID * 2);
    __bf16*   pw2l   = (__bf16*)carve((size_t)HID * HID * 2);
    __bf16*   pw2r   = (__bf16*)carve((size_t)HID * HID * 2);
    __bf16*   pw3l   = (__bf16*)carve((size_t)HID * HID * 2);
    __bf16*   pw3r   = (__bf16*)carve((size_t)HID * HID * 2);
    __bf16*   plin1  = (__bf16*)carve((size_t)HID * HID * 2);
    unsigned* mkey   = (unsigned*)carve((size_t)n_graphs * sizeof(unsigned));
    float*    ssum   = (float*)carve((size_t)n_graphs * sizeof(float));
    float*    pnum   = (float*)carve((size_t)n_graphs * HID * sizeof(float));
    float*    pooled = (float*)carve((size_t)n_graphs * HID * sizeof(float));
    float*    h1mlp  = (float*)carve((size_t)n_graphs * HID * sizeof(float));
    (void)ws_size; (void)n_in;

    const int T = 256;
    auto blocksLL = [](long long n, int t) { return (unsigned)((n + t - 1) / t); };

    // ---- degrees (shared across layers) ----
    zero_f32<<<blocksLL(n_nodes, T), T, 0, stream>>>((float*)cnt, n_nodes);
    count_deg<<<blocksLL(n_edges, T), T, 0, stream>>>(ei, cnt, n_edges);
    inv_deg<<<blocksLL(n_nodes, T), T, 0, stream>>>(cnt, inv, n_nodes);

    // ---- pack weights into WMMA B fragments (bf16) ----
    pack_b<<<4, T, 0, stream>>>(w1l, pw1l);
    pack_b<<<4, T, 0, stream>>>(w1r, pw1r);
    pack_b<<<4, T, 0, stream>>>(w2l, pw2l);
    pack_b<<<4, T, 0, stream>>>(w2r, pw2r);
    pack_b<<<4, T, 0, stream>>>(w3l, pw3l);
    pack_b<<<4, T, 0, stream>>>(w3r, pw3r);
    pack_b<<<4, T, 0, stream>>>(lin1_w, plin1);

    const long long nodeElems  = (long long)n_nodes * HID;
    const long long edgeThr    = (long long)n_edges * 32;
    const long long nodeThr    = (long long)n_nodes * 32;
    const unsigned  gemmBlocks = (unsigned)(((n_nodes / 16) + 3) / 4);

    // ---- layer 1: x -> hA ----
    zero_f32<<<blocksLL(nodeElems, T), T, 0, stream>>>(msg, nodeElems);
    scatter_edges<<<blocksLL(edgeThr, T), T, 0, stream>>>(x, ei, msg, n_edges);
    sage_gemm<true><<<gemmBlocks, 128, 0, stream>>>(msg, inv, x, pw1l, pw1r, b1, hA, n_nodes, 1);

    // ---- layer 2: hA -> hB ----
    zero_f32<<<blocksLL(nodeElems, T), T, 0, stream>>>(msg, nodeElems);
    scatter_edges<<<blocksLL(edgeThr, T), T, 0, stream>>>(hA, ei, msg, n_edges);
    sage_gemm<true><<<gemmBlocks, 128, 0, stream>>>(msg, inv, hA, pw2l, pw2r, b2, hB, n_nodes, 1);

    // ---- layer 3: hB -> hA ----
    zero_f32<<<blocksLL(nodeElems, T), T, 0, stream>>>(msg, nodeElems);
    scatter_edges<<<blocksLL(edgeThr, T), T, 0, stream>>>(hB, ei, msg, n_edges);
    sage_gemm<true><<<gemmBlocks, 128, 0, stream>>>(msg, inv, hB, pw3l, pw3r, b3, hA, n_nodes, 1);

    // ---- attention pooling ----
    zero_f32<<<blocksLL(n_graphs, T), T, 0, stream>>>((float*)mkey, n_graphs);
    zero_f32<<<blocksLL(n_graphs, T), T, 0, stream>>>(ssum, n_graphs);
    zero_f32<<<blocksLL((long long)n_graphs * HID, T), T, 0, stream>>>(pnum, (long long)n_graphs * HID);
    gate_kernel<<<blocksLL(nodeThr, T), T, 0, stream>>>(hA, gate_w, gate_b, batch, gate, mkey, n_nodes);
    att_accum<<<blocksLL(nodeThr, T), T, 0, stream>>>(hA, gate, batch, mkey, ssum, pnum, n_nodes);
    pooled_div<<<blocksLL((long long)n_graphs * HID, T), T, 0, stream>>>(pnum, ssum, pooled, n_graphs * HID);

    // ---- MLP head: lin1 via WMMA, lin2 + log_softmax scalar ----
    const unsigned gemmBlocksG = (unsigned)(((n_graphs / 16) + 3) / 4);
    sage_gemm<false><<<gemmBlocksG, 128, 0, stream>>>(pooled, nullptr, nullptr, plin1, nullptr,
                                                      lin1_b, h1mlp, n_graphs, 1);
    head_kernel<<<n_graphs, HID, 0, stream>>>(h1mlp, lin2_w, lin2_b, (float*)d_out, n_classes);
}
